// VisionTransformer_42288247996982
// MI455X (gfx1250) — matrix operand order repaired and checked by reading, not compiled, not run on hardware
//
#include <hip/hip_runtime.h>
#include <math.h>
#include <stdint.h>

// ---------------- model constants ----------------
#define BATCH   64
#define CH      3
#define IMG     224
#define PATCH   16
#define DIM     768
#define DEPTH   12
#define HEADS   12
#define DHEAD   64          // DIM / HEADS
#define NPATCH  196         // (224/16)^2
#define SEQ     197         // NPATCH + 1
#define SP      224         // padded sequence (multiple of 32)
#define MP      (BATCH * SP)   // 14336 padded token rows
#define FF      3072
#define LNEPS   1e-6f

typedef __attribute__((ext_vector_type(4)))  __bf16 v4bf;
typedef __attribute__((ext_vector_type(8)))  float  v8f;
typedef __attribute__((ext_vector_type(8)))  __bf16 v8bf;
typedef __attribute__((ext_vector_type(16))) __bf16 v16bf;

#define BF16_CAT(lo, hi) __builtin_shufflevector((lo), (hi), \
    0,1,2,3,4,5,6,7,8,9,10,11,12,13,14,15)

// ---------------- epilogue ids ----------------
#define EPI_F32       0   // fp32 store (+bias if non-null)
#define EPI_BF16      1   // bf16 store (+bias if non-null)
#define EPI_BF16_GELU 2   // bf16 store, bias + exact GELU
#define EPI_F32_RES   3   // fp32 store, bias + residual add (in place OK)

// =====================================================================
// Generic batched NT WMMA GEMM.
//   A: [batch] x (M x K) bf16 row-major        (stride sA elements / batch)
//   B: [batch] x (N x K) bf16 row-major        (stride sB elements / batch)
//   C: element (m,n) at coff + m*ldc + n, coff = (z/cdiv)*sC1 + (z%cdiv)*sC2
// Block: 256 threads = 8 waves -> 256(M) x 128(N); wave tile 64x64
// (16 accumulators -> 16 WMMA per 16 b128-loads per K-step, 32 flop/B).
// No LDS/barriers: out-of-range waves exit wave-uniformly; EXEC is all-1
// through the WMMA loop (ISA requirement); store guards handle ragged
// attention tiles (divergence only in the epilogue).
// =====================================================================
template <int EPI>
__global__ __launch_bounds__(256) void gemm_nt_wmma(
    const __bf16* __restrict__ A, const __bf16* __restrict__ B,
    const float* __restrict__ bias, const float* res,
    float* Cf, __bf16* Cb,
    int M, int N, int K,
    long long sA, long long sB, long long sC1, long long sC2,
    int cdiv, int ldc)
{
    const int lane = threadIdx.x & 31;
    const int wave = threadIdx.x >> 5;
    const int m0 = blockIdx.x * 256 + (wave >> 1) * 64;
    const int n0 = blockIdx.y * 128 + (wave & 1) * 64;
    if (m0 >= M || n0 >= N) return;           // wave-uniform exit

    const int z = blockIdx.z;
    A += (size_t)z * (size_t)sA;
    B += (size_t)z * (size_t)sB;
    const size_t coff = (size_t)(z / cdiv) * (size_t)sC1 +
                        (size_t)(z % cdiv) * (size_t)sC2;

    const int lm  = lane & 15;
    const int alo = (lane & 16) ? 8 : 0;    // A: lanes 16-31 hold K+8 / K+24
    const int bko = (lane & 16) ? 8 : 0;    // B: lanes 16-31 hold K+8 / K+24

    const __bf16* ap[4];
    const __bf16* bp[4];
#pragma unroll
    for (int i = 0; i < 4; ++i) {
        ap[i] = A + (size_t)(m0 + 16 * i + lm) * K;
        bp[i] = B + (size_t)(n0 + 16 * i + lm) * K + bko;
    }

    v8f acc[4][4] = {};

    for (int k0 = 0; k0 < K; k0 += 32) {
        if (k0 + 32 < K) {                      // global_prefetch_b8 path
            __builtin_prefetch((const void*)(bp[0] + k0 + 32), 0, 1);
            __builtin_prefetch((const void*)(ap[0] + k0 + 32), 0, 1);
        }
        v16bf fa[4], fb[4];
#pragma unroll
        for (int i = 0; i < 4; ++i) {
            v8bf lo = *(const v8bf*)(ap[i] + k0 + alo);
            v8bf hi = *(const v8bf*)(ap[i] + k0 + 16 + alo);
            fa[i] = BF16_CAT(lo, hi);
            fb[i] = BF16_CAT(*(const v8bf*)(bp[i] + k0), *(const v8bf*)(bp[i] + k0 + 16));
        }
#pragma unroll
        for (int i = 0; i < 4; ++i)
#pragma unroll
            for (int j = 0; j < 4; ++j)
                acc[i][j] = __builtin_amdgcn_wmma_f32_16x16x32_bf16(
                    false, fa[i], false, fb[j], (short)0, acc[i][j],
                    false, false);
    }

    // C/D layout: VGPR r, lane l -> row = tile_m + r + 8*(l>=16), col = tile_n + l%16
    const int hi8 = (lane >> 4) * 8;
#pragma unroll
    for (int i = 0; i < 4; ++i) {
#pragma unroll
        for (int j = 0; j < 4; ++j) {
#pragma unroll
            for (int r = 0; r < 8; ++r) {
                const int row = m0 + 16 * i + hi8 + r;
                const int col = n0 + 16 * j + lm;
                if (row >= M || col >= N) continue;
                const size_t idx = coff + (size_t)row * ldc + col;
                float v = acc[i][j][r];
                if (bias) v += bias[col];
                if constexpr (EPI == EPI_F32) {
                    Cf[idx] = v;
                } else if constexpr (EPI == EPI_BF16) {
                    Cb[idx] = (__bf16)v;
                } else if constexpr (EPI == EPI_BF16_GELU) {
                    float g = 0.5f * v * (1.0f + erff(v * 0.70710678118654752f));
                    Cb[idx] = (__bf16)g;
                } else { // EPI_F32_RES
                    Cf[idx] = v + res[idx];
                }
            }
        }
    }
}

// =====================================================================
// Elementwise / reshape / normalization kernels
// =====================================================================
__global__ void cvt_f32_bf16(const float* __restrict__ in,
                             __bf16* __restrict__ out, int n)
{
    int i = blockIdx.x * 256 + threadIdx.x;
    if (i < n) out[i] = (__bf16)in[i];
}

// in: [R x Cc] fp32, out: [Cc x R] bf16  (weight [K,N] -> [N,K])
__global__ void transpose_f32_bf16(const float* __restrict__ in,
                                   __bf16* __restrict__ out, int R, int Cc)
{
    int i = blockIdx.x * 256 + threadIdx.x;
    if (i >= R * Cc) return;
    int c = i / R, r = i - c * R;
    out[i] = (__bf16)in[(size_t)r * Cc + c];
}

// Build im2col patch matrix: xcol[MP x 768] bf16; rows s=0 and s>=197 are 0.
__global__ void im2col_patches(const float* __restrict__ x,
                               __bf16* __restrict__ xcol)
{
    int idx = blockIdx.x * 256 + threadIdx.x;
    if (idx >= MP * DIM) return;
    int m = idx / DIM, k = idx - m * DIM;
    int b = m / SP, s = m - b * SP;
    float v = 0.f;
    if (s >= 1 && s <= NPATCH) {
        int p = s - 1;
        int py = p / 14, px = p - py * 14;
        int c = k >> 8, rem = k & 255, ii = rem >> 4, jj = rem & 15;
        v = x[(((size_t)b * CH + c) * IMG + (py * PATCH + ii)) * IMG +
              (px * PATCH + jj)];
    }
    xcol[idx] = (__bf16)v;
}

// h already holds conv result + patch bias. Set cls row, add pos, zero pads.
__global__ void embed_finish(float* __restrict__ h,
                             const float* __restrict__ cls,
                             const float* __restrict__ pos)
{
    int idx = blockIdx.x * 256 + threadIdx.x;
    if (idx >= MP * DIM) return;
    int m = idx / DIM, d = idx - m * DIM;
    int s = m % SP;
    if (s == 0)            h[idx] = cls[d] + pos[d];
    else if (s < SEQ)      h[idx] = h[idx] + pos[(size_t)s * DIM + d];
    else                   h[idx] = 0.f;
}

// LayerNorm per row of 768, fp32 in -> bf16 out. One wave per row.
// float4 loads (b128), packed 4xbf16 stores (b64).
__global__ __launch_bounds__(256) void ln_rows(
    const float* __restrict__ x, const float* __restrict__ g,
    const float* __restrict__ bta, __bf16* __restrict__ y, int nrows)
{
    const int lane = threadIdx.x & 31, wave = threadIdx.x >> 5;
    const int row = blockIdx.x * 8 + wave;
    if (row >= nrows) return;
    const float4* xr = (const float4*)(x + (size_t)row * DIM);
    float v[24], s = 0.f;
#pragma unroll
    for (int i = 0; i < 6; ++i) {
        float4 t = xr[i * 32 + lane];
        v[4*i+0] = t.x; v[4*i+1] = t.y; v[4*i+2] = t.z; v[4*i+3] = t.w;
        s += t.x + t.y + t.z + t.w;
    }
    for (int m = 16; m; m >>= 1) s += __shfl_xor(s, m, 32);
    const float mu = s * (1.f / DIM);
    float q = 0.f;
#pragma unroll
    for (int i = 0; i < 24; ++i) { float d = v[i] - mu; q += d * d; }
    for (int m = 16; m; m >>= 1) q += __shfl_xor(q, m, 32);
    const float rs = rsqrtf(q * (1.f / DIM) + LNEPS);
    const float4* g4 = (const float4*)g;
    const float4* b4 = (const float4*)bta;
    v4bf* yr = (v4bf*)(y + (size_t)row * DIM);
#pragma unroll
    for (int i = 0; i < 6; ++i) {
        float4 gv = g4[i * 32 + lane];
        float4 bv = b4[i * 32 + lane];
        v4bf o;
        o[0] = (__bf16)((v[4*i+0] - mu) * rs * gv.x + bv.x);
        o[1] = (__bf16)((v[4*i+1] - mu) * rs * gv.y + bv.y);
        o[2] = (__bf16)((v[4*i+2] - mu) * rs * gv.z + bv.z);
        o[3] = (__bf16)((v[4*i+3] - mu) * rs * gv.w + bv.w);
        yr[i * 32 + lane] = o;
    }
}

// Final LN on s==0 rows only -> fp32 output [B, 768].
__global__ __launch_bounds__(256) void ln_head(
    const float* __restrict__ x, const float* __restrict__ g,
    const float* __restrict__ bta, float* __restrict__ out)
{
    const int lane = threadIdx.x & 31, wave = threadIdx.x >> 5;
    const int b = blockIdx.x * 8 + wave;
    if (b >= BATCH) return;
    const float4* xr = (const float4*)(x + (size_t)b * SP * DIM);  // s==0 row
    float v[24], s = 0.f;
#pragma unroll
    for (int i = 0; i < 6; ++i) {
        float4 t = xr[i * 32 + lane];
        v[4*i+0] = t.x; v[4*i+1] = t.y; v[4*i+2] = t.z; v[4*i+3] = t.w;
        s += t.x + t.y + t.z + t.w;
    }
    for (int m = 16; m; m >>= 1) s += __shfl_xor(s, m, 32);
    const float mu = s * (1.f / DIM);
    float q = 0.f;
#pragma unroll
    for (int i = 0; i < 24; ++i) { float d = v[i] - mu; q += d * d; }
    for (int m = 16; m; m >>= 1) q += __shfl_xor(q, m, 32);
    const float rs = rsqrtf(q * (1.f / DIM) + LNEPS);
    const float4* g4 = (const float4*)g;
    const float4* b4 = (const float4*)bta;
    float4* orow = (float4*)(out + (size_t)b * DIM);
#pragma unroll
    for (int i = 0; i < 6; ++i) {
        float4 gv = g4[i * 32 + lane];
        float4 bv = b4[i * 32 + lane];
        float4 o;
        o.x = (v[4*i+0] - mu) * rs * gv.x + bv.x;
        o.y = (v[4*i+1] - mu) * rs * gv.y + bv.y;
        o.z = (v[4*i+2] - mu) * rs * gv.z + bv.z;
        o.w = (v[4*i+3] - mu) * rs * gv.w + bv.w;
        orow[i * 32 + lane] = o;
    }
}

// Split heads: q/k/v [MP x 768] bf16 ->
//   Qh [B*H, SP, 64] (Q scaled by 1/8), Kh [B*H, SP, 64], Vt [B*H, 64, SP]
__global__ void reshape_qkv(const __bf16* __restrict__ q,
                            const __bf16* __restrict__ k,
                            const __bf16* __restrict__ v,
                            __bf16* __restrict__ Qh,
                            __bf16* __restrict__ Kh,
                            __bf16* __restrict__ Vt)
{
    int idx = blockIdx.x * 256 + threadIdx.x;
    if (idx >= MP * DIM) return;
    int m = idx / DIM, n = idx - m * DIM;
    int b = m / SP, s = m - b * SP;
    int hh = n >> 6, d = n & 63;
    int z = b * HEADS + hh;
    size_t hd = ((size_t)z * SP + s) * DHEAD + d;
    Qh[hd] = (__bf16)((float)q[idx] * 0.125f);   // fold 1/sqrt(64)
    Kh[hd] = k[idx];
    Vt[((size_t)z * DHEAD + d) * SP + s] = v[idx];
}

// Masked softmax over t (valid t < 197) for rows of SP=224. Wave per row.
__global__ __launch_bounds__(256) void softmax_rows(
    const float* __restrict__ sc, __bf16* __restrict__ pr, int nrows)
{
    const int lane = threadIdx.x & 31, wave = threadIdx.x >> 5;
    const int row = blockIdx.x * 8 + wave;
    if (row >= nrows) return;
    const size_t base = (size_t)row * SP;
    float v[7], mx = -1e30f;
#pragma unroll
    for (int i = 0; i < 7; ++i) {
        int t = i * 32 + lane;
        float x = (t < SEQ) ? sc[base + t] : -1e30f;
        v[i] = x; mx = fmaxf(mx, x);
    }
    for (int m = 16; m; m >>= 1) mx = fmaxf(mx, __shfl_xor(mx, m, 32));
    float s = 0.f;
#pragma unroll
    for (int i = 0; i < 7; ++i) {
        int t = i * 32 + lane;
        v[i] = (t < SEQ) ? expf(v[i] - mx) : 0.f;
        s += v[i];
    }
    for (int m = 16; m; m >>= 1) s += __shfl_xor(s, m, 32);
    const float inv = 1.f / s;
#pragma unroll
    for (int i = 0; i < 7; ++i)
        pr[base + i * 32 + lane] = (__bf16)(v[i] * inv);
}

// =====================================================================
// Host side
// =====================================================================
static inline int ceil_div(int a, int b) { return (a + b - 1) / b; }

static void launch_gemm(int epi,
    const __bf16* A, const __bf16* B, const float* bias, const float* res,
    float* Cf, __bf16* Cb, int M, int N, int K,
    long long sA, long long sB, long long sC1, long long sC2,
    int cdiv, int ldc, int batch, hipStream_t st)
{
    dim3 grid(ceil_div(M, 256), ceil_div(N, 128), batch), blk(256);
    switch (epi) {
    case EPI_F32:
        gemm_nt_wmma<EPI_F32><<<grid, blk, 0, st>>>(A, B, bias, res, Cf, Cb,
            M, N, K, sA, sB, sC1, sC2, cdiv, ldc); break;
    case EPI_BF16:
        gemm_nt_wmma<EPI_BF16><<<grid, blk, 0, st>>>(A, B, bias, res, Cf, Cb,
            M, N, K, sA, sB, sC1, sC2, cdiv, ldc); break;
    case EPI_BF16_GELU:
        gemm_nt_wmma<EPI_BF16_GELU><<<grid, blk, 0, st>>>(A, B, bias, res, Cf, Cb,
            M, N, K, sA, sB, sC1, sC2, cdiv, ldc); break;
    default:
        gemm_nt_wmma<EPI_F32_RES><<<grid, blk, 0, st>>>(A, B, bias, res, Cf, Cb,
            M, N, K, sA, sB, sC1, sC2, cdiv, ldc); break;
    }
}

extern "C" void kernel_launch(void* const* d_in, const int* in_sizes, int n_in,
                              void* d_out, int out_size, void* d_ws, size_t ws_size,
                              hipStream_t stream)
{
    (void)in_sizes; (void)n_in; (void)out_size; (void)ws_size;

    const float* x_in    = (const float*)d_in[0];
    const float* patch_w = (const float*)d_in[1];
    const float* patch_b = (const float*)d_in[2];
    const float* cls_tok = (const float*)d_in[3];
    const float* pos_emb = (const float*)d_in[4];
    const float* ln1_s   = (const float*)d_in[5];
    const float* ln1_b   = (const float*)d_in[6];
    const float* qw      = (const float*)d_in[7];
    const float* qb      = (const float*)d_in[8];
    const float* kw      = (const float*)d_in[9];
    const float* kb      = (const float*)d_in[10];
    const float* vw      = (const float*)d_in[11];
    const float* vb      = (const float*)d_in[12];
    const float* pw      = (const float*)d_in[13];
    const float* pb      = (const float*)d_in[14];
    const float* ln2_s   = (const float*)d_in[15];
    const float* ln2_b   = (const float*)d_in[16];
    const float* fc1_w   = (const float*)d_in[17];
    const float* fc1_b   = (const float*)d_in[18];
    const float* fc2_w   = (const float*)d_in[19];
    const float* fc2_b   = (const float*)d_in[20];
    const float* lnf_s   = (const float*)d_in[21];
    const float* lnf_b   = (const float*)d_in[22];
    float* out = (float*)d_out;

    // ---- workspace carve-out (256B aligned) ----
    char* wp_ = (char*)d_ws;
    auto take = [&](size_t bytes) -> char* {
        char* r = wp_;
        wp_ += (bytes + 255) & ~(size_t)255;
        return r;
    };
    float*  h      = (float*) take((size_t)MP * DIM * 4);
    __bf16* ybf    = (__bf16*)take((size_t)MP * DIM * 2);   // LN out / xcol
    __bf16* qbuf   = (__bf16*)take((size_t)MP * DIM * 2);   // also obuf
    __bf16* kbuf   = (__bf16*)take((size_t)MP * DIM * 2);
    __bf16* vbuf   = (__bf16*)take((size_t)MP * DIM * 2);
    __bf16* Qh     = (__bf16*)take((size_t)BATCH * HEADS * SP * DHEAD * 2);
    __bf16* Kh     = (__bf16*)take((size_t)BATCH * HEADS * SP * DHEAD * 2);
    __bf16* Vt     = (__bf16*)take((size_t)BATCH * HEADS * DHEAD * SP * 2);
    float*  scores = (float*) take((size_t)BATCH * HEADS * SP * SP * 4);
    __bf16* probs  = (__bf16*)take((size_t)BATCH * HEADS * SP * SP * 2);
    __bf16* wqT    = (__bf16*)take((size_t)DIM * DIM * 2);  // also patch_w bf16
    __bf16* wkT    = (__bf16*)take((size_t)DIM * DIM * 2);
    __bf16* wvT    = (__bf16*)take((size_t)DIM * DIM * 2);
    __bf16* wpT    = (__bf16*)take((size_t)DIM * DIM * 2);
    __bf16* fc1T   = (__bf16*)take((size_t)DIM * FF * 2);
    __bf16* fc2T   = (__bf16*)take((size_t)FF * DIM * 2);
    __bf16* mlp1   = (__bf16*)scores;   // alias: scores dead before MLP
    __bf16* obuf   = qbuf;              // alias: qbuf dead after head split

    const int ELT = MP * DIM;                 // 11,010,048
    dim3 blk(256);

    // ---- patch embedding: im2col GEMM ----
    cvt_f32_bf16<<<ceil_div(DIM * DIM, 256), blk, 0, stream>>>(
        patch_w, wqT, DIM * DIM);             // [768,768] already [N,K]
    im2col_patches<<<ceil_div(ELT, 256), blk, 0, stream>>>(x_in, ybf);
    launch_gemm(EPI_F32, ybf, wqT, patch_b, nullptr, h, nullptr,
                MP, DIM, DIM, 0, 0, 0, 0, 1, DIM, 1, stream);
    embed_finish<<<ceil_div(ELT, 256), blk, 0, stream>>>(h, cls_tok, pos_emb);

    const long long sHD = (long long)SP * DHEAD;   // per (b,h) Q/K stride
    const long long sSS = (long long)SP * SP;

    for (int l = 0; l < DEPTH; ++l) {
        const size_t wo = (size_t)l * DIM * DIM;
        // weight transposes -> [N,K] bf16
        transpose_f32_bf16<<<ceil_div(DIM * DIM, 256), blk, 0, stream>>>(
            qw + wo, wqT, DIM, DIM);
        transpose_f32_bf16<<<ceil_div(DIM * DIM, 256), blk, 0, stream>>>(
            kw + wo, wkT, DIM, DIM);
        transpose_f32_bf16<<<ceil_div(DIM * DIM, 256), blk, 0, stream>>>(
            vw + wo, wvT, DIM, DIM);
        transpose_f32_bf16<<<ceil_div(DIM * DIM, 256), blk, 0, stream>>>(
            pw + wo, wpT, DIM, DIM);
        transpose_f32_bf16<<<ceil_div(DIM * FF, 256), blk, 0, stream>>>(
            fc1_w + (size_t)l * DIM * FF, fc1T, DIM, FF);
        transpose_f32_bf16<<<ceil_div(FF * DIM, 256), blk, 0, stream>>>(
            fc2_w + (size_t)l * FF * DIM, fc2T, FF, DIM);

        // ---- attention ----
        ln_rows<<<ceil_div(MP, 8), blk, 0, stream>>>(
            h, ln1_s + l * DIM, ln1_b + l * DIM, ybf, MP);
        launch_gemm(EPI_BF16, ybf, wqT, qb + l * DIM, nullptr, nullptr, qbuf,
                    MP, DIM, DIM, 0, 0, 0, 0, 1, DIM, 1, stream);
        launch_gemm(EPI_BF16, ybf, wkT, kb + l * DIM, nullptr, nullptr, kbuf,
                    MP, DIM, DIM, 0, 0, 0, 0, 1, DIM, 1, stream);
        launch_gemm(EPI_BF16, ybf, wvT, vb + l * DIM, nullptr, nullptr, vbuf,
                    MP, DIM, DIM, 0, 0, 0, 0, 1, DIM, 1, stream);
        reshape_qkv<<<ceil_div(ELT, 256), blk, 0, stream>>>(
            qbuf, kbuf, vbuf, Qh, Kh, Vt);

        // scores[z] = Qh[z] (SPx64) x Kh[z]^T : batched NT, K=64
        launch_gemm(EPI_F32, Qh, Kh, nullptr, nullptr, scores, nullptr,
                    SP, SP, DHEAD, sHD, sHD, sSS, 0, 1, SP,
                    BATCH * HEADS, stream);
        softmax_rows<<<ceil_div(BATCH * HEADS * SP, 8), blk, 0, stream>>>(
            scores, probs, BATCH * HEADS * SP);
        // O[z] = probs[z] (SPxSP) x V[z] : B = Vt [64 x SP]; C -> [MP x 768]
        launch_gemm(EPI_BF16, probs, Vt, nullptr, nullptr, nullptr, obuf,
                    SP, DHEAD, SP, sSS, sHD, (long long)SP * DIM, DHEAD,
                    HEADS, DIM, BATCH * HEADS, stream);
        // h += O x Wp^T + pb
        launch_gemm(EPI_F32_RES, obuf, wpT, pb + l * DIM, h, h, nullptr,
                    MP, DIM, DIM, 0, 0, 0, 0, 1, DIM, 1, stream);

        // ---- MLP ----
        ln_rows<<<ceil_div(MP, 8), blk, 0, stream>>>(
            h, ln2_s + l * DIM, ln2_b + l * DIM, ybf, MP);
        launch_gemm(EPI_BF16_GELU, ybf, fc1T, fc1_b + (size_t)l * FF, nullptr,
                    nullptr, mlp1, MP, FF, DIM, 0, 0, 0, 0, 1, FF, 1, stream);
        launch_gemm(EPI_F32_RES, mlp1, fc2T, fc2_b + l * DIM, h, h, nullptr,
                    MP, DIM, FF, 0, 0, 0, 0, 1, DIM, 1, stream);
    }

    // ---- final LN on cls rows -> output [64, 768] fp32 ----
    ln_head<<<ceil_div(BATCH, 8), blk, 0, stream>>>(h, lnf_s, lnf_b, out);
}
